// EdgeAwareAttentionGNN_10823317586521
// MI455X (gfx1250) — matrix operand order repaired and checked by reading, not compiled, not run on hardware
//
#include <hip/hip_runtime.h>
#include <hip/hip_bf16.h>
#include <math.h>

typedef __attribute__((ext_vector_type(16))) __bf16 v16bf;
typedef __attribute__((ext_vector_type(8)))  float  v8f;
typedef unsigned short ushort_t;
typedef unsigned int   uint_t;

__device__ __forceinline__ ushort_t f2bf_bits(float f) {
  // round-to-nearest-even f32 -> bf16 (bit level)
  unsigned u = __float_as_uint(f);
  unsigned r = u + 0x7FFFu + ((u >> 16) & 1u);
  return (ushort_t)(r >> 16);
}

__device__ __forceinline__ void atomicMaxF(float* addr, float val) {
  if (val >= 0.0f) atomicMax((int*)addr, __float_as_int(val));
  else             atomicMin((unsigned int*)addr, __float_as_uint(val));
}

__device__ __forceinline__ void reduce8_xor(float v[8]) {
  #pragma unroll
  for (int mask = 1; mask <= 8; mask <<= 1) {
    #pragma unroll
    for (int r = 0; r < 8; ++r) v[r] += __shfl_xor(v[r], mask, 32);
  }
}

// ---------------------------------------------------------------------------
__global__ void fill_kernel(float* p, float v, int n) {
  int i = blockIdx.x * blockDim.x + threadIdx.x;
  if (i < n) p[i] = v;
}

// f32[n] -> packed bf16[n] (n even)
__global__ void cvt_bf16_kernel(const float* __restrict__ in, ushort_t* __restrict__ out, int n) {
  int i = (blockIdx.x * blockDim.x + threadIdx.x) * 2;
  if (i >= n) return;
  uint_t u = (uint_t)f2bf_bits(in[i]) | ((uint_t)f2bf_bits(in[i + 1]) << 16);
  *(uint_t*)(out + i) = u;
}

// W[K,64] f32 -> Wt[64,K] bf16 (transposed, packed pairs along K)
__global__ void cvt_wt_kernel(const float* __restrict__ W, ushort_t* __restrict__ Wt, int K) {
  int tid = blockIdx.x * blockDim.x + threadIdx.x;
  int total = 64 * (K / 2);
  if (tid >= total) return;
  int n = tid / (K / 2);
  int k = (tid % (K / 2)) * 2;
  uint_t u = (uint_t)f2bf_bits(W[(size_t)k * 64 + n]) |
             ((uint_t)f2bf_bits(W[(size_t)(k + 1) * 64 + n]) << 16);
  *(uint_t*)(Wt + (size_t)n * K + k) = u;
}

// ---------------------------------------------------------------------------
// M[j,h] = sum_c We[j, h*C+c] * ae[h,c]
// ---------------------------------------------------------------------------
__global__ void precontract_kernel(const float* __restrict__ We1, const float* __restrict__ ae1,
                                   const float* __restrict__ We2, const float* __restrict__ ae2,
                                   float* __restrict__ M1, float* __restrict__ M2) {
  int t = threadIdx.x;
  if (t < 64) {
    int j = t >> 2, hh = t & 3;
    float s = 0.f;
    for (int c = 0; c < 16; ++c) s += We1[j * 64 + hh * 16 + c] * ae1[hh * 16 + c];
    M1[j * 4 + hh] = s;
  }
  if (t < 16) {
    float s = 0.f;
    for (int c = 0; c < 64; ++c) s += We2[t * 64 + c] * ae2[c];
    M2[t] = s;
  }
}

__global__ void deg_loop_kernel(const int* __restrict__ dst, const float* __restrict__ ea,
                                float* __restrict__ deg, float* __restrict__ loopat, int E) {
  int e = blockIdx.x * blockDim.x + threadIdx.x;
  if (e >= E) return;
  int d = dst[e];
  atomicAdd(&deg[d], 1.0f);
  const float4* v = (const float4*)(ea + (size_t)e * 16);
  float* lp = loopat + (size_t)d * 16;
  #pragma unroll
  for (int q = 0; q < 4; ++q) {
    float4 t = v[q];
    atomicAdd(&lp[q * 4 + 0], t.x);
    atomicAdd(&lp[q * 4 + 1], t.y);
    atomicAdd(&lp[q * 4 + 2], t.z);
    atomicAdd(&lp[q * 4 + 3], t.w);
  }
}

__global__ void loopnorm_kernel(const float* __restrict__ deg, const float* __restrict__ loopat,
                                const float* __restrict__ M1, const float* __restrict__ M2,
                                float* __restrict__ aleL1, float* __restrict__ aleL2, int N) {
  int n = blockIdx.x * blockDim.x + threadIdx.x;
  if (n >= N) return;
  float dinv = 1.0f / fmaxf(deg[n], 1.0f);
  float la[16];
  #pragma unroll
  for (int j = 0; j < 16; ++j) la[j] = loopat[(size_t)n * 16 + j] * dinv;
  #pragma unroll
  for (int hh = 0; hh < 4; ++hh) {
    float s = 0.f;
    #pragma unroll
    for (int j = 0; j < 16; ++j) s += la[j] * M1[j * 4 + hh];
    aleL1[(size_t)n * 4 + hh] = s;
  }
  float s2 = 0.f;
  #pragma unroll
  for (int j = 0; j < 16; ++j) s2 += la[j] * M2[j];
  aleL2[n] = s2;
}

// ---------------------------------------------------------------------------
// WMMA bf16 GEMM: Hout[N,64] = Xbf[N,K] @ Wt^T   (Wt is bf16 [64,K])
// Wt staged in LDS (shared across 4 waves/block). One wave = one 16-row tile.
// A/B fragments are contiguous 16B runs matching ISA §7.12.2 layouts.
// Fused epilogue: als/ald attention logits via xor-shuffle reductions.
// ---------------------------------------------------------------------------
template <int K, int H>
__global__ void gat_gemm_wmma(const ushort_t* __restrict__ Xbf, const ushort_t* __restrict__ Wt,
                              float* __restrict__ Hout, float* __restrict__ als,
                              float* __restrict__ ald,
                              const float* __restrict__ avs, const float* __restrict__ avd,
                              int N) {
  __shared__ ushort_t sW[64 * K];
  {
    const uint_t* s = (const uint_t*)Wt;
    uint_t* d = (uint_t*)sW;
    for (int i = threadIdx.x; i < 64 * K / 2; i += blockDim.x) d[i] = s[i];
  }
  __syncthreads();

  const int lane = threadIdx.x & 31;
  const int wave = threadIdx.x >> 5;
  const int tile = blockIdx.x * (blockDim.x >> 5) + wave;
  const int row0 = tile * 16;
  if (row0 >= N) return;  // wave-uniform: EXEC all-ones for WMMA
  const int half = lane >> 4;
  const int lm   = lane & 15;

  v8f acc[4] = {};
  const int m = (row0 + lm < N) ? (row0 + lm) : (N - 1);
  const ushort_t* xr = Xbf + (size_t)m * K;

  #pragma unroll
  for (int kb = 0; kb < K; kb += 32) {
    // A fragment: half 0 -> K {kb..kb+7, kb+16..kb+23}; half 1 -> +8
    union { uint4 u[2]; v16bf v; } ca;
    ca.u[0] = *(const uint4*)(xr + kb + half * 8);
    ca.u[1] = *(const uint4*)(xr + kb + 16 + half * 8);
    #pragma unroll
    for (int t = 0; t < 4; ++t) {
      // B fragment: column t*16+lm, K {kb..kb+7, kb+16..kb+23} + half*8
      const ushort_t* wr = sW + (size_t)(t * 16 + lm) * K + kb + half * 8;
      union { uint4 u[2]; v16bf v; } cb;
      cb.u[0] = *(const uint4*)(wr);
      cb.u[1] = *(const uint4*)(wr + 16);
      acc[t] = __builtin_amdgcn_wmma_f32_16x16x32_bf16(
          false, ca.v, false, cb.v, (short)0, acc[t], false, false);
    }
  }

  // Store D: lane holds column lm, rows r + 8*half
  #pragma unroll
  for (int t = 0; t < 4; ++t) {
    #pragma unroll
    for (int r = 0; r < 8; ++r) {
      int row = row0 + r + 8 * half;
      if (row < N) Hout[(size_t)row * 64 + t * 16 + lm] = acc[t][r];
    }
  }

  // Fused attention-logit reductions (xor shuffles stay within 16-lane half)
  if (H == 4) {
    #pragma unroll
    for (int t = 0; t < 4; ++t) {
      float ps[8], pd[8];
      float cs = avs[t * 16 + lm], cd = avd[t * 16 + lm];
      #pragma unroll
      for (int r = 0; r < 8; ++r) { float hv = acc[t][r]; ps[r] = hv * cs; pd[r] = hv * cd; }
      reduce8_xor(ps);
      reduce8_xor(pd);
      if (lm == 0) {
        #pragma unroll
        for (int r = 0; r < 8; ++r) {
          int row = row0 + r + 8 * half;
          if (row < N) { als[(size_t)row * 4 + t] = ps[r]; ald[(size_t)row * 4 + t] = pd[r]; }
        }
      }
    }
  } else {
    float ps[8] = {0, 0, 0, 0, 0, 0, 0, 0};
    float pd[8] = {0, 0, 0, 0, 0, 0, 0, 0};
    #pragma unroll
    for (int t = 0; t < 4; ++t) {
      float cs = avs[t * 16 + lm], cd = avd[t * 16 + lm];
      #pragma unroll
      for (int r = 0; r < 8; ++r) { float hv = acc[t][r]; ps[r] += hv * cs; pd[r] += hv * cd; }
    }
    reduce8_xor(ps);
    reduce8_xor(pd);
    if (lm == 0) {
      #pragma unroll
      for (int r = 0; r < 8; ++r) {
        int row = row0 + r + 8 * half;
        if (row < N) { als[row] = ps[r]; ald[row] = pd[r]; }
      }
    }
  }
}

// ---------------------------------------------------------------------------
// Softmax pass 1: alpha = LeakyReLU(als[src]+ald[dst]+al_e) ; segment max
// ---------------------------------------------------------------------------
__global__ void attn_alpha_max_kernel(const int* __restrict__ src, const int* __restrict__ dst,
                                      const float* __restrict__ ea, const float* __restrict__ M,
                                      const float* __restrict__ als, const float* __restrict__ ald,
                                      const float* __restrict__ aleL,
                                      float* __restrict__ alpha, float* __restrict__ alphaL,
                                      float* __restrict__ amax, int E, int N, int H) {
  int i = blockIdx.x * blockDim.x + threadIdx.x;
  if (i < E) {
    int s = src[i], d = dst[i];
    const float* eav = ea + (size_t)i * 16;
    for (int h = 0; h < H; ++h) {
      float al = 0.f;
      #pragma unroll
      for (int j = 0; j < 16; ++j) al += eav[j] * M[j * H + h];
      float a = als[(size_t)s * H + h] + ald[(size_t)d * H + h] + al;
      a = (a > 0.f) ? a : 0.2f * a;
      alpha[(size_t)i * H + h] = a;
      atomicMaxF(&amax[(size_t)d * H + h], a);
    }
  } else if (i < E + N) {
    int n = i - E;
    for (int h = 0; h < H; ++h) {
      float a = als[(size_t)n * H + h] + ald[(size_t)n * H + h] + aleL[(size_t)n * H + h];
      a = (a > 0.f) ? a : 0.2f * a;
      alphaL[(size_t)n * H + h] = a;
      atomicMaxF(&amax[(size_t)n * H + h], a);
    }
  }
}

__global__ void attn_expsum_kernel(const int* __restrict__ dst,
                                   float* __restrict__ alpha, float* __restrict__ alphaL,
                                   const float* __restrict__ amax, float* __restrict__ den,
                                   int E, int N, int H) {
  int i = blockIdx.x * blockDim.x + threadIdx.x;
  if (i < E) {
    int d = dst[i];
    for (int h = 0; h < H; ++h) {
      float ex = __expf(alpha[(size_t)i * H + h] - amax[(size_t)d * H + h]);
      alpha[(size_t)i * H + h] = ex;
      atomicAdd(&den[(size_t)d * H + h], ex);
    }
  } else if (i < E + N) {
    int n = i - E;
    for (int h = 0; h < H; ++h) {
      float ex = __expf(alphaL[(size_t)n * H + h] - amax[(size_t)n * H + h]);
      alphaL[(size_t)n * H + h] = ex;
      atomicAdd(&den[(size_t)n * H + h], ex);
    }
  }
}

__global__ void attn_agg_kernel(const int* __restrict__ src, const int* __restrict__ dst,
                                const float* __restrict__ alpha, const float* __restrict__ alphaL,
                                const float* __restrict__ den, const float* __restrict__ Hf,
                                float* __restrict__ outacc, int E, int N, int H) {
  int tid = blockIdx.x * blockDim.x + threadIdx.x;
  int id = tid >> 2, chunk = tid & 3;
  if (id >= E + N) return;
  int head = chunk & (H - 1);  // H in {1,4}
  int s, d;
  float att;
  if (id < E) {
    s = src[id]; d = dst[id];
    att = alpha[(size_t)id * H + head] / den[(size_t)d * H + head];
  } else {
    int n = id - E;
    s = n; d = n;
    att = alphaL[(size_t)n * H + head] / den[(size_t)n * H + head];
  }
  const float4* hv = (const float4*)(Hf + (size_t)s * 64 + chunk * 16);
  float* op = outacc + (size_t)d * 64 + chunk * 16;
  #pragma unroll
  for (int q = 0; q < 4; ++q) {
    float4 v = hv[q];
    atomicAdd(&op[q * 4 + 0], v.x * att);
    atomicAdd(&op[q * 4 + 1], v.y * att);
    atomicAdd(&op[q * 4 + 2], v.z * att);
    atomicAdd(&op[q * 4 + 3], v.w * att);
  }
}

__global__ void bias_elu_kernel(float* __restrict__ acc, const float* __restrict__ b, int total) {
  int i = blockIdx.x * blockDim.x + threadIdx.x;
  if (i >= total) return;
  float v = acc[i] + b[i & 63];
  acc[i] = (v > 0.f) ? v : (__expf(v) - 1.f);
}

__global__ void pool_kernel(const float* __restrict__ acc2, const int* __restrict__ batch,
                            float* __restrict__ pooled, float* __restrict__ cnt, int N) {
  int tid = blockIdx.x * blockDim.x + threadIdx.x;
  int n = tid >> 2, chunk = tid & 3;
  if (n >= N) return;
  int g = batch[n];
  const float4* v = (const float4*)(acc2 + (size_t)n * 64 + chunk * 16);
  float* pp = pooled + (size_t)g * 64 + chunk * 16;
  #pragma unroll
  for (int q = 0; q < 4; ++q) {
    float4 t = v[q];
    atomicAdd(&pp[q * 4 + 0], t.x);
    atomicAdd(&pp[q * 4 + 1], t.y);
    atomicAdd(&pp[q * 4 + 2], t.z);
    atomicAdd(&pp[q * 4 + 3], t.w);
  }
  if (chunk == 0) atomicAdd(&cnt[g], 1.0f);
}

__global__ void final_proj_kernel(const float* __restrict__ pooled, const float* __restrict__ cnt,
                                  const float* __restrict__ b2, const float* __restrict__ Pw,
                                  const float* __restrict__ Pb, float* __restrict__ out, int G) {
  int tid = threadIdx.x;
  int total = G * 64;
  for (int o = tid; o < total; o += blockDim.x) {
    int g = o >> 6, c = o & 63;
    float cn = fmaxf(cnt[g], 1.0f);
    float s = Pb[c];
    for (int j = 0; j < 64; ++j)
      s += (pooled[(size_t)g * 64 + j] / cn + b2[j]) * Pw[(size_t)j * 64 + c];
    out[o] = s;
  }
}

// ---------------------------------------------------------------------------
// Host launcher
// ---------------------------------------------------------------------------
extern "C" void kernel_launch(void* const* d_in, const int* in_sizes, int n_in,
                              void* d_out, int out_size, void* d_ws, size_t ws_size,
                              hipStream_t stream) {
  (void)n_in; (void)ws_size;
  const float* x    = (const float*)d_in[0];
  const int*   ei   = (const int*)d_in[1];
  const float* ea   = (const float*)d_in[2];
  const int*   batch= (const int*)d_in[3];
  const float* W1   = (const float*)d_in[4];
  const float* as1  = (const float*)d_in[5];
  const float* ad1  = (const float*)d_in[6];
  const float* We1  = (const float*)d_in[7];
  const float* ae1  = (const float*)d_in[8];
  const float* b1   = (const float*)d_in[9];
  const float* W2   = (const float*)d_in[10];
  const float* as2  = (const float*)d_in[11];
  const float* ad2  = (const float*)d_in[12];
  const float* We2  = (const float*)d_in[13];
  const float* ae2  = (const float*)d_in[14];
  const float* b2   = (const float*)d_in[15];
  const float* Pw   = (const float*)d_in[16];
  const float* Pb   = (const float*)d_in[17];

  const int N = in_sizes[0] / 128;
  const int E = in_sizes[1] / 2;
  const int G = out_size / 64;
  const int* srcp = ei;
  const int* dstp = ei + E;

  char* ws = (char*)d_ws;
  size_t off = 0;
  auto carve = [&](size_t bytes) -> void* {
    void* p = (void*)(ws + off);
    off += (bytes + 255) & ~(size_t)255;
    return p;
  };
  ushort_t* Xbf = (ushort_t*)carve((size_t)N * 128 * 2);  // bf16 X (layer2 reuses N*64)
  ushort_t* Wt1 = (ushort_t*)carve(64 * 128 * 2);
  ushort_t* Wt2 = (ushort_t*)carve(64 * 64 * 2);
  float* hbuf   = (float*)carve((size_t)N * 64 * 4);      // h1, later h2
  float* accA   = (float*)carve((size_t)N * 64 * 4);      // layer1 agg -> ELU -> layer2 input
  float* accB   = (float*)carve((size_t)N * 64 * 4);      // layer2 agg
  float* als    = (float*)carve((size_t)N * 4 * 4);
  float* ald    = (float*)carve((size_t)N * 4 * 4);
  float* deg    = (float*)carve((size_t)N * 4);
  float* loopat = (float*)carve((size_t)N * 16 * 4);
  float* aleL1  = (float*)carve((size_t)N * 4 * 4);
  float* aleL2  = (float*)carve((size_t)N * 4);
  float* alpha  = (float*)carve((size_t)E * 4 * 4);
  float* alphaL = (float*)carve((size_t)N * 4 * 4);
  float* amax   = (float*)carve((size_t)N * 4 * 4);
  float* den    = (float*)carve((size_t)N * 4 * 4);
  float* M1     = (float*)carve(64 * 4);
  float* M2     = (float*)carve(16 * 4);
  float* pooled = (float*)carve((size_t)G * 64 * 4);
  float* cnt    = (float*)carve((size_t)G * 4);

  const int TB = 256;
  auto nb = [](long n, int tb) { return (unsigned)((n + tb - 1) / tb); };
  const float NEG_INF = -__builtin_inff();

  // ---- init (re-done every call: deterministic under graph replay) ----
  fill_kernel<<<nb((long)N, TB), TB, 0, stream>>>(deg, 0.f, N);
  fill_kernel<<<nb((long)N * 16, TB), TB, 0, stream>>>(loopat, 0.f, N * 16);
  fill_kernel<<<nb((long)N * 4, TB), TB, 0, stream>>>(den, 0.f, N * 4);
  fill_kernel<<<nb((long)N * 4, TB), TB, 0, stream>>>(amax, NEG_INF, N * 4);
  fill_kernel<<<nb((long)N * 64, TB), TB, 0, stream>>>(accA, 0.f, N * 64);
  fill_kernel<<<nb((long)N * 64, TB), TB, 0, stream>>>(accB, 0.f, N * 64);
  fill_kernel<<<nb((long)G * 64, TB), TB, 0, stream>>>(pooled, 0.f, G * 64);
  fill_kernel<<<nb((long)G, TB), TB, 0, stream>>>(cnt, 0.f, G);

  // ---- one-time conversions / precontractions ----
  cvt_bf16_kernel<<<nb((long)N * 128 / 2, TB), TB, 0, stream>>>(x, Xbf, N * 128);
  cvt_wt_kernel<<<nb(64 * 128 / 2, TB), TB, 0, stream>>>(W1, Wt1, 128);
  cvt_wt_kernel<<<nb(64 * 64 / 2, TB), TB, 0, stream>>>(W2, Wt2, 64);
  precontract_kernel<<<1, 64, 0, stream>>>(We1, ae1, We2, ae2, M1, M2);
  deg_loop_kernel<<<nb(E, TB), TB, 0, stream>>>(dstp, ea, deg, loopat, E);
  loopnorm_kernel<<<nb(N, TB), TB, 0, stream>>>(deg, loopat, M1, M2, aleL1, aleL2, N);

  const int tiles = (N + 15) / 16;

  // ---- layer 1 ----
  gat_gemm_wmma<128, 4><<<(tiles + 3) / 4, 128, 0, stream>>>(Xbf, Wt1, hbuf, als, ald, as1, ad1, N);
  attn_alpha_max_kernel<<<nb((long)E + N, TB), TB, 0, stream>>>(srcp, dstp, ea, M1, als, ald,
                                                                aleL1, alpha, alphaL, amax, E, N, 4);
  attn_expsum_kernel<<<nb((long)E + N, TB), TB, 0, stream>>>(dstp, alpha, alphaL, amax, den, E, N, 4);
  attn_agg_kernel<<<nb(((long)E + N) * 4, TB), TB, 0, stream>>>(srcp, dstp, alpha, alphaL, den,
                                                                hbuf, accA, E, N, 4);
  bias_elu_kernel<<<nb((long)N * 64, TB), TB, 0, stream>>>(accA, b1, N * 64);

  // ---- layer 2 (H=1; reuse softmax buffers after refill) ----
  fill_kernel<<<nb((long)N, TB), TB, 0, stream>>>(den, 0.f, N);
  fill_kernel<<<nb((long)N, TB), TB, 0, stream>>>(amax, NEG_INF, N);
  cvt_bf16_kernel<<<nb((long)N * 64 / 2, TB), TB, 0, stream>>>(accA, Xbf, N * 64);
  gat_gemm_wmma<64, 1><<<(tiles + 3) / 4, 128, 0, stream>>>(Xbf, Wt2, hbuf, als, ald, as2, ad2, N);
  attn_alpha_max_kernel<<<nb((long)E + N, TB), TB, 0, stream>>>(srcp, dstp, ea, M2, als, ald,
                                                                aleL2, alpha, alphaL, amax, E, N, 1);
  attn_expsum_kernel<<<nb((long)E + N, TB), TB, 0, stream>>>(dstp, alpha, alphaL, amax, den, E, N, 1);
  attn_agg_kernel<<<nb(((long)E + N) * 4, TB), TB, 0, stream>>>(srcp, dstp, alpha, alphaL, den,
                                                                hbuf, accB, E, N, 1);

  // ---- pool + project ----
  pool_kernel<<<nb((long)N * 4, TB), TB, 0, stream>>>(accB, batch, pooled, cnt, N);
  final_proj_kernel<<<1, 256, 0, stream>>>(pooled, cnt, b2, Pw, Pb, (float*)d_out, G);
}
